// modeler_30709016166544
// MI455X (gfx1250) — matrix operand order repaired and checked by reading, not compiled, not run on hardware
//
#include <hip/hip_runtime.h>
#include <math.h>

#define NV   50000
#define NU   20000
#define DEG  16
#define FT   128
#define HID  256
#define OUT_ 128

typedef __attribute__((ext_vector_type(16))) __bf16 v16bf;
typedef __attribute__((ext_vector_type(8)))  __bf16 v8bf;
typedef __attribute__((ext_vector_type(4)))  __bf16 v4bf;
typedef __attribute__((ext_vector_type(8)))  float  v8f;
typedef __attribute__((ext_vector_type(4)))  float  f32x4;

__device__ __forceinline__ void splitf(float x, __bf16& h, __bf16& l) {
    h = (__bf16)x;
    l = (__bf16)(x - (float)h);
}

// ---------------------------------------------------------------------------
// Weight prep: W [K,N] fp32 row-major  ->  Wt_hi/Wt_lo [N,K] bf16 (transposed)
// so a B-fragment (16 contiguous k at fixed n) is one aligned 32B load.
// ---------------------------------------------------------------------------
__global__ void __launch_bounds__(256)
prep_w(const float* __restrict__ W, int K, int N,
       __bf16* __restrict__ Wh, __bf16* __restrict__ Wl)
{
    const int idx = blockIdx.x * 256 + threadIdx.x;   // over K*N, k fastest
    if (idx >= K * N) return;
    const int n = idx / K, k = idx % K;
    __bf16 h, l; splitf(W[k * N + n], h, l);
    Wh[n * K + k] = h;
    Wl[n * K + k] = l;
}

// ---------------------------------------------------------------------------
// Neighbor-mean: one wave per node, lane = one 16B chunk (global_load_b128).
// grid = M/4 blocks of 128 threads.  CH = K/128 float4 chunks per lane.
// ---------------------------------------------------------------------------
template<int K>
__global__ void __launch_bounds__(128)
agg_mean4(const float* __restrict__ feat, const int* __restrict__ nbr,
          float* __restrict__ out, float scale)
{
    constexpr int CH = K / 128;
    const int wave = threadIdx.x >> 5, lane = threadIdx.x & 31;
    const int i = blockIdx.x * 4 + wave;
    const int* nb = nbr + i * DEG;
    f32x4 acc[CH];
    #pragma unroll
    for (int c = 0; c < CH; ++c) acc[c] = (f32x4){0.f, 0.f, 0.f, 0.f};
    #pragma unroll
    for (int j = 0; j < DEG; ++j) {
        if (j + 1 < DEG)
            __builtin_prefetch(feat + (size_t)nb[j + 1] * K + lane * 4, 0, 1);
        const f32x4* row = (const f32x4*)(feat + (size_t)nb[j] * K);
        #pragma unroll
        for (int c = 0; c < CH; ++c) acc[c] += row[lane + 32 * c];
    }
    f32x4* o = (f32x4*)(out + (size_t)i * K);
    #pragma unroll
    for (int c = 0; c < CH; ++c) o[lane + 32 * c] = acc[c] * scale;
}

// ---------------------------------------------------------------------------
// sv[i,:] = scale * (sum_j other[nbr[i,j],:] + self[i,:])   (K = 128)
// ---------------------------------------------------------------------------
__global__ void __launch_bounds__(128)
sum_self4(const float* __restrict__ other, const int* __restrict__ nbr,
          const float* __restrict__ self_, float* __restrict__ out, float scale)
{
    const int wave = threadIdx.x >> 5, lane = threadIdx.x & 31;
    const int i = blockIdx.x * 4 + wave;
    const int* nb = nbr + i * DEG;
    f32x4 acc = ((const f32x4*)(self_ + (size_t)i * OUT_))[lane];
    #pragma unroll
    for (int j = 0; j < DEG; ++j)
        acc += ((const f32x4*)(other + (size_t)nb[j] * OUT_))[lane];
    ((f32x4*)(out + (size_t)i * OUT_))[lane] = acc * scale;
}

// ---------------------------------------------------------------------------
// Fused GEMM + bias + optional PReLU, bf16x3 split (Ah*Bh + Al*Bh + Ah*Bl),
// fp32 accumulate via v_wmma_f32_16x16x32_bf16.
//   X: split-K concat view (k < Ka -> Xa, else Xb), staged to LDS as bf16 hi/lo.
//   Wh/Wl: prepacked [NDIM, KTOT] bf16 transposed weights.
//   Block: 16 rows x NDIM, 4 waves, TILES 16-wide N-tiles per wave.
// ---------------------------------------------------------------------------
template<int KTOT, int NDIM>
__global__ void __launch_bounds__(128)
gemm_bias_act(const float* __restrict__ Xa, int Ka,
              const float* __restrict__ Xb,
              const __bf16* __restrict__ Wh, const __bf16* __restrict__ Wl,
              const float* __restrict__ bias,
              const float* __restrict__ alpha,   // null => identity
              float* __restrict__ Y, int M)
{
    static_assert(KTOT % 32 == 0 && NDIM % 64 == 0, "tile shape");
    constexpr int LDK   = KTOT + 8;      // keeps 16B alignment (8 | LDK)
    constexpr int TILES = NDIM / 16 / 4;

    __shared__ __bf16 Ah[16 * LDK];
    __shared__ __bf16 Al[16 * LDK];

    const int row0 = blockIdx.x * 16;
    const int tid  = threadIdx.x;

    // Stage 16 x KTOT tile: float4 global loads -> packed bf16 hi/lo (b64 DS stores)
    constexpr int KC = KTOT / 4;
    for (int idx = tid; idx < 16 * KC; idx += 128) {
        const int r = idx / KC, k = (idx % KC) * 4;
        const int grow = row0 + r;
        const f32x4 x = (k < Ka)
            ? *(const f32x4*)(Xa + (size_t)grow * Ka + k)
            : *(const f32x4*)(Xb + (size_t)grow * (KTOT - Ka) + (k - Ka));
        v4bf hv, lv;
        #pragma unroll
        for (int e = 0; e < 4; ++e) {
            __bf16 h, l; splitf(x[e], h, l);
            hv[e] = h; lv[e] = l;
        }
        *(v4bf*)&Ah[r * LDK + k] = hv;
        *(v4bf*)&Al[r * LDK + k] = lv;
    }
    __syncthreads();

    const int lane = tid & 31;
    const int wave = tid >> 5;
    const int m    = lane & 15;
    const int hh   = lane >> 4;

    v8f c[TILES];
    #pragma unroll
    for (int t = 0; t < TILES; ++t)
        c[t] = (v8f){0.f, 0.f, 0.f, 0.f, 0.f, 0.f, 0.f, 0.f};

    for (int kk = 0; kk < KTOT; kk += 32) {
        // A fragment = two contiguous 16B runs per lane (ds_load_b128 x2)
        const int abase = m * LDK + kk + 16 * hh;
        const v8bf a0h = *(const v8bf*)&Ah[abase];
        const v8bf a1h = *(const v8bf*)&Ah[abase + 8];
        const v8bf a0l = *(const v8bf*)&Al[abase];
        const v8bf a1l = *(const v8bf*)&Al[abase + 8];
        const v16bf ah = __builtin_shufflevector(a0h, a1h,
            0, 1, 2, 3, 4, 5, 6, 7, 8, 9, 10, 11, 12, 13, 14, 15);
        const v16bf al = __builtin_shufflevector(a0l, a1l,
            0, 1, 2, 3, 4, 5, 6, 7, 8, 9, 10, 11, 12, 13, 14, 15);
        #pragma unroll
        for (int t = 0; t < TILES; ++t) {
            const int n = (wave * TILES + t) * 16 + (lane & 15);
            // B fragment = 16 contiguous bf16 at fixed n (32B aligned load)
            const int bbase = n * KTOT + kk + 16 * hh;
            const v16bf bh = *(const v16bf*)&Wh[bbase];
            const v16bf bl = *(const v16bf*)&Wl[bbase];
            c[t] = __builtin_amdgcn_wmma_f32_16x16x32_bf16(
                       false, ah, false, bh, (short)0, c[t], false, false);
            c[t] = __builtin_amdgcn_wmma_f32_16x16x32_bf16(
                       false, al, false, bh, (short)0, c[t], false, false);
            c[t] = __builtin_amdgcn_wmma_f32_16x16x32_bf16(
                       false, ah, false, bl, (short)0, c[t], false, false);
        }
    }

    const float a = alpha ? alpha[0] : 0.f;
    #pragma unroll
    for (int t = 0; t < TILES; ++t) {
        const int n = (wave * TILES + t) * 16 + (lane & 15);
        const float b = bias[n];
        #pragma unroll
        for (int r = 0; r < 8; ++r) {         // C/D layout: M = r + 8*hh
            float y = c[t][r] + b;
            if (alpha) y = (y >= 0.f) ? y : a * y;
            Y[(size_t)(row0 + r + 8 * hh) * NDIM + n] = y;
        }
    }
}

// ---------------------------------------------------------------------------
// Hinge-loss partials (deterministic fixed-order reduction, float4 dots).
// ---------------------------------------------------------------------------
__global__ void __launch_bounds__(256)
loss_partial(const float* __restrict__ e3, const float* __restrict__ s,
             const int* __restrict__ perm, int M, float* __restrict__ part)
{
    const int wave = threadIdx.x >> 5;
    const int lane = threadIdx.x & 31;
    float acc = 0.f;
    for (int i = blockIdx.x * 8 + wave; i < M; i += gridDim.x * 8) {
        const int j = perm[i];
        const f32x4 e  = ((const f32x4*)(e3 + (size_t)i * OUT_))[lane];
        const f32x4 sp = ((const f32x4*)(s  + (size_t)i * OUT_))[lane];
        const f32x4 sn = ((const f32x4*)(s  + (size_t)j * OUT_))[lane];
        const f32x4 tp = e * sp, tn = e * sn;
        float p = tp.x + tp.y + tp.z + tp.w;
        float n = tn.x + tn.y + tn.z + tn.w;
        #pragma unroll
        for (int off = 16; off; off >>= 1) {
            p += __shfl_xor(p, off);
            n += __shfl_xor(n, off);
        }
        if (lane == 0) {
            const float vpl = 1.f / (1.f + expf(-p));
            const float vnl = 1.f / (1.f + expf(-n));
            const float h = vnl - vpl + 0.5f;
            acc += (h > 0.f) ? h : 0.f;
        }
    }
    __shared__ float wsum[8];
    if (lane == 0) wsum[wave] = acc;
    __syncthreads();
    if (threadIdx.x == 0) {
        float tot = 0.f;
        #pragma unroll
        for (int w = 0; w < 8; ++w) tot += wsum[w];
        part[blockIdx.x] = tot;
    }
}

__global__ void loss_final(const float* __restrict__ pv, const float* __restrict__ pu,
                           float* __restrict__ out)
{
    float sv = 0.f, su = 0.f;
    for (int i = 0; i < 512; ++i) sv += pv[i];
    for (int i = 0; i < 512; ++i) su += pu[i];
    out[0] = sv / (float)NV + su / (float)NU;
}

// ---------------------------------------------------------------------------
extern "C" void kernel_launch(void* const* d_in, const int* in_sizes, int n_in,
                              void* d_out, int out_size, void* d_ws, size_t ws_size,
                              hipStream_t stream)
{
    const float* feat_v = (const float*)d_in[0];
    const float* feat_u = (const float*)d_in[1];
    const int*   nbr_v  = (const int*)d_in[2];
    const int*   nbr_u  = (const int*)d_in[3];
    const int*   vidx   = (const int*)d_in[4];
    const int*   uidx   = (const int*)d_in[5];
    const float* W1v = (const float*)d_in[6],  *b1v = (const float*)d_in[7];
    const float* W1u = (const float*)d_in[8],  *b1u = (const float*)d_in[9];
    const float* a1v = (const float*)d_in[10], *a1u = (const float*)d_in[11];
    const float* W2v = (const float*)d_in[12], *b2v = (const float*)d_in[13];
    const float* W2u = (const float*)d_in[14], *b2u = (const float*)d_in[15];
    const float* a2v = (const float*)d_in[16], *a2u = (const float*)d_in[17];
    const float* W3v = (const float*)d_in[18], *b3v = (const float*)d_in[19];
    const float* W3u = (const float*)d_in[20], *b3u = (const float*)d_in[21];

    // Workspace layout (floats), regions reused as lifetimes end:
    float* ws    = (float*)d_ws;
    float* ve1   = ws;                                   // NV*256 : ve1 -> ve3
    float* ue1   = ve1  + (size_t)NV * HID;              // NU*256 : ue1 -> ue3
    float* vagg  = ue1  + (size_t)NU * HID;              // NV*256 : vn -> vagg2 -> sv
    float* uagg  = vagg + (size_t)NV * HID;              // NU*256 : un -> uagg2 -> su
    float* partV = uagg + (size_t)NU * HID;              // 512
    float* partU = partV + 512;                          // 512

    // Prepacked bf16 hi/lo transposed weights (12 arrays of 32768 = FT*HID)
    __bf16* wp = (__bf16*)(partU + 512);
    const size_t WSZ = (size_t)FT * HID;                 // == HID*OUT_ == 32768
    __bf16 *W1v_h = wp,            *W1v_l = wp + WSZ;
    __bf16 *W1u_h = wp + 2 * WSZ,  *W1u_l = wp + 3 * WSZ;
    __bf16 *W2v_h = wp + 4 * WSZ,  *W2v_l = wp + 5 * WSZ;
    __bf16 *W2u_h = wp + 6 * WSZ,  *W2u_l = wp + 7 * WSZ;
    __bf16 *W3v_h = wp + 8 * WSZ,  *W3v_l = wp + 9 * WSZ;
    __bf16 *W3u_h = wp + 10 * WSZ, *W3u_l = wp + 11 * WSZ;

    float* ve2  = (float*)d_out;                         // NV*128
    float* ue2  = ve2 + (size_t)NV * OUT_;               // NU*128
    float* loss = ue2 + (size_t)NU * OUT_;               // 1

    float* ve3 = ve1;  float* ue3 = ue1;
    float* sv  = vagg; float* su  = uagg;
    const float inv_deg = 1.f / (float)DEG;
    const float inv_dp1 = 1.f / (float)(DEG + 1);
    const int PB = (int)((WSZ + 255) / 256);

    // Weight prepack (reads only d_in; independent of the pipeline below)
    prep_w<<<PB, 256, 0, stream>>>(W1v, FT,  HID,  W1v_h, W1v_l);
    prep_w<<<PB, 256, 0, stream>>>(W1u, FT,  HID,  W1u_h, W1u_l);
    prep_w<<<PB, 256, 0, stream>>>(W2v, HID, OUT_, W2v_h, W2v_l);
    prep_w<<<PB, 256, 0, stream>>>(W2u, HID, OUT_, W2u_h, W2u_l);
    prep_w<<<PB, 256, 0, stream>>>(W3v, HID, OUT_, W3v_h, W3v_l);
    prep_w<<<PB, 256, 0, stream>>>(W3u, HID, OUT_, W3u_h, W3u_l);

    // Layer 1 aggregation + GEMM(+PReLU)
    agg_mean4<FT><<<NV / 4, 128, 0, stream>>>(feat_u, nbr_v, vagg, inv_deg);
    agg_mean4<FT><<<NU / 4, 128, 0, stream>>>(feat_v, nbr_u, uagg, inv_deg);
    gemm_bias_act<FT, HID><<<NV / 16, 128, 0, stream>>>(vagg, FT, vagg, W1v_h, W1v_l, b1v, a1v, ve1, NV);
    gemm_bias_act<FT, HID><<<NU / 16, 128, 0, stream>>>(uagg, FT, uagg, W1u_h, W1u_l, b1u, a1u, ue1, NU);

    // Layer 2 aggregation (overwrites vn/un) + GEMM(+PReLU) -> d_out
    agg_mean4<HID><<<NV / 4, 128, 0, stream>>>(ue1, nbr_v, vagg, inv_deg);
    agg_mean4<HID><<<NU / 4, 128, 0, stream>>>(ve1, nbr_u, uagg, inv_deg);
    gemm_bias_act<HID, OUT_><<<NV / 16, 128, 0, stream>>>(vagg, HID, vagg, W2v_h, W2v_l, b2v, a2v, ve2, NV);
    gemm_bias_act<HID, OUT_><<<NU / 16, 128, 0, stream>>>(uagg, HID, uagg, W2u_h, W2u_l, b2u, a2u, ue2, NU);

    // Layer 3: concat(ve2, feat) @ W3 + b3, identity (split-K concat view)
    gemm_bias_act<HID, OUT_><<<NV / 16, 128, 0, stream>>>(ve2, OUT_, feat_v, W3v_h, W3v_l, b3v, nullptr, ve3, NV);
    gemm_bias_act<HID, OUT_><<<NU / 16, 128, 0, stream>>>(ue2, OUT_, feat_u, W3u_h, W3u_l, b3u, nullptr, ue3, NU);

    // sv / su
    sum_self4<<<NV / 4, 128, 0, stream>>>(ue3, nbr_v, ve3, sv, inv_dp1);
    sum_self4<<<NU / 4, 128, 0, stream>>>(ve3, nbr_u, ue3, su, inv_dp1);

    // Loss (deterministic two-stage reduction)
    loss_partial<<<512, 256, 0, stream>>>(ve3, sv, vidx, NV, partV);
    loss_partial<<<512, 256, 0, stream>>>(ue3, su, uidx, NU, partU);
    loss_final<<<1, 1, 0, stream>>>(partV, partU, loss);
}
